// LSTMTimeDecoder_66520453481162
// MI455X (gfx1250) — compile-verified, matrix operand order checked by reading, not run on hardware
//
#include <hip/hip_runtime.h>
#include <hip/hip_bf16.h>

// ---------------------------------------------------------------------------
// LSTM time decoder for MI455X (gfx1250, wave32, WMMA).
//   H=512 hidden, O=128 out, S=512 steps, B=256 batch.
//
// Decomposition (exact algebra, see analysis):
//   gates[s,b,g] = base2[b,g] + t_delta[s,b]*u[g] + (h[s-1] @ W_hh^T)[b,g]
//   base2 = C @ W_ih[:, :H]^T + time_b @ W_ih[:, H:]^T + b_ih + b_hh
//   u     = W_ih[:, H:] @ time_W
// Recurrence runs in ONE persistent kernel (64 WGs) with a global step
// barrier; W_hh tile in LDS, cell state c + base2 fragments in registers,
// gate GEMM via v_wmma_f32_16x16x32_bf16. Final linear is a parallel WMMA
// GEMM over all S*B rows.
// ---------------------------------------------------------------------------

#define HD   512
#define OD   128
#define SL   512
#define BA   256
#define G4H  (4 * HD)          // 2048
#define NWG  64                // persistent grid for recurrence
#define LP   528               // LDS row pitch in bf16 (512 + 16 pad)

typedef __attribute__((ext_vector_type(16))) __bf16 v16bf;
typedef __attribute__((ext_vector_type(8)))  float  v8f;

extern __shared__ char smem_raw[];

__device__ __forceinline__ float sigf(float x) { return 1.0f / (1.0f + __expf(-x)); }
__device__ __forceinline__ float tanh_fast(float x) {
  float e = __expf(2.0f * x);
  return 1.0f - 2.0f / (e + 1.0f);
}

// ---------------- setup kernels (one-time, cheap) ----------------

__global__ void zero_h0(__bf16* hsx, unsigned int* bar) {
  int i = blockIdx.x * blockDim.x + threadIdx.x;
  if (i < BA * HD) hsx[i] = (__bf16)0.0f;
  if (i == 0) *bar = 0u;
}

__global__ void compute_tdelta(const float* __restrict__ t, float* __restrict__ tdel) {
  int i = blockIdx.x * blockDim.x + threadIdx.x;   // s*B + b
  if (i >= SL * BA) return;
  tdel[i] = (i < BA) ? 0.0f : (t[i] - t[i - BA]);
}

__global__ void compute_uv(const float* __restrict__ W_ih, const float* __restrict__ b_ih,
                           const float* __restrict__ b_hh, const float* __restrict__ time_W,
                           const float* __restrict__ time_b,
                           float* __restrict__ u, float* __restrict__ v2) {
  int g = blockIdx.x * blockDim.x + threadIdx.x;
  if (g >= G4H) return;
  const float* row = W_ih + (size_t)g * (2 * HD) + HD;   // time half of W_ih row g
  float su = 0.0f, sv = 0.0f;
  for (int h = 0; h < HD; ++h) { float w = row[h]; su += time_W[h] * w; sv += time_b[h] * w; }
  u[g]  = su;
  v2[g] = sv + b_ih[g] + b_hh[g];
}

__global__ void convert_bf16(const float* __restrict__ W_hh, const float* __restrict__ lin_W,
                             __bf16* __restrict__ whhb, __bf16* __restrict__ linwb) {
  int i = blockIdx.x * blockDim.x + threadIdx.x;
  if (i < G4H * HD) whhb[i] = (__bf16)W_hh[i];
  else {
    int j = i - G4H * HD;
    if (j < OD * HD) linwb[j] = (__bf16)lin_W[j];
  }
}

__global__ void compute_base2(const float* __restrict__ C, const float* __restrict__ W_ih,
                              const float* __restrict__ v2, float* __restrict__ base2) {
  int g = blockIdx.x * blockDim.x + threadIdx.x;   // 0..2047
  int b = blockIdx.y;                              // 0..255
  const float* crow = C + (size_t)b * HD;
  const float* wrow = W_ih + (size_t)g * (2 * HD); // C half of W_ih row g
  float s = 0.0f;
  for (int h = 0; h < HD; ++h) s += crow[h] * wrow[h];
  base2[(size_t)b * G4H + g] = s + v2[g];
}

// ---------------- persistent recurrent kernel ----------------
// 64 WGs x 256 threads (8 wave32). WG (b0,h0) owns a 64x32 (batch x hidden)
// tile of h/c. Wave w: mt=w&3 (16 batch rows), nt=w>>2 (16 hidden cols);
// computes the 4 gate tiles i,f,g,o for that (b,h) 16x16 tile each step.

__global__ __launch_bounds__(256) void lstm_persistent(
    const float* __restrict__ tdel,    // [S,B]
    const float* __restrict__ base2,   // [B,4H]
    const float* __restrict__ u,       // [4H]
    const __bf16* __restrict__ whhb,   // [4H,H] row-major
    __bf16* __restrict__ hsx,          // [(S+1),B,H]; row 0 zeroed
    unsigned int* bar)
{
  __bf16* lds = (__bf16*)smem_raw;     // 128 rows x LP halves (W_hh slice)

  const int wg  = blockIdx.x;
  const int b0  = (wg & 3) * 64;       // batch tile origin
  const int h0  = (wg >> 2) * 32;      // hidden tile origin
  const int tid = threadIdx.x;
  const int w   = tid >> 5, lane = tid & 31;
  const int mt  = w & 3, nt = w >> 2;
  const int n   = lane & 15;           // fragment column
  const int hi  = lane >> 4;           // K-half / M-half select
  const int gr  = b0 + mt * 16;        // wave's global batch-row base
  const int gc  = h0 + nt * 16;        // wave's global hidden-col base

  // ---- stage W_hh slice into LDS once: 4 gates x 32 rows, 512 K each ----
  for (int idx = tid; idx < 128 * 64; idx += 256) {
    int row = idx >> 6, ch = idx & 63; // 8 halves per chunk
    int g = (row >> 5) * HD + h0 + (row & 31);
    *(uint4*)(lds + row * LP + ch * 8) =
        *(const uint4*)(whhb + (size_t)g * HD + ch * 8);
  }
  __syncthreads();

  // ---- persistent per-wave state ----
  v8f   base_f[4];
  float u_f[4];
  #pragma unroll
  for (int gate = 0; gate < 4; ++gate) {
    u_f[gate] = u[gate * HD + gc + n];
    #pragma unroll
    for (int r = 0; r < 8; ++r)
      base_f[gate][r] = base2[(size_t)(gr + r + 8 * hi) * G4H + gate * HD + gc + n];
  }
  v8f c_acc = {};                      // cell state lives in registers

  for (int s = 0; s < SL; ++s) {
    // time deltas for this wave's 8 (m = r + 8*hi) rows
    float td[8];
    {
      const float* tp = tdel + (size_t)s * BA + gr + 8 * hi;
      float4 t0 = *(const float4*)(tp);
      float4 t1 = *(const float4*)(tp + 4);
      td[0] = t0.x; td[1] = t0.y; td[2] = t0.z; td[3] = t0.w;
      td[4] = t1.x; td[5] = t1.y; td[6] = t1.z; td[7] = t1.w;
    }

    v8f acc[4];
    #pragma unroll
    for (int gate = 0; gate < 4; ++gate)
      #pragma unroll
      for (int r = 0; r < 8; ++r)
        acc[gate][r] = base_f[gate][r] + td[r] * u_f[gate];

    const __bf16* hprev = hsx + (size_t)s * (BA * HD);
    #pragma unroll 4
    for (int kk = 0; kk < HD / 32; ++kk) {
      int kbase = kk * 32 + hi * 16;
      v16bf a = *(const v16bf*)(hprev + (size_t)(gr + n) * HD + kbase);
      #pragma unroll
      for (int gate = 0; gate < 4; ++gate) {
        v16bf bf = *(const v16bf*)(lds + (gate * 32 + nt * 16 + n) * LP + kbase);
        acc[gate] = __builtin_amdgcn_wmma_f32_16x16x32_bf16(
            false, a, false, bf, (short)0, acc[gate], false, false);
      }
    }

    // LSTM cell update (register-local) + publish h_t
    __bf16* hout = hsx + (size_t)(s + 1) * (BA * HD);
    #pragma unroll
    for (int r = 0; r < 8; ++r) {
      float ig = sigf(acc[0][r]);
      float fg = sigf(acc[1][r]);
      float gg = tanh_fast(acc[2][r]);
      float og = sigf(acc[3][r]);
      float c  = fg * c_acc[r] + ig * gg;
      c_acc[r] = c;
      float hv = og * tanh_fast(c);
      hout[(size_t)(gr + r + 8 * hi) * HD + gc + n] = (__bf16)hv;
    }

    // ---- device-wide step barrier (monotonic counter, 64 resident WGs) ----
    __threadfence();
    __syncthreads();
    if (tid == 0) {
      atomicAdd(bar, 1u);
      unsigned tgt = (unsigned)NWG * (unsigned)(s + 1);
      while (__hip_atomic_load(bar, __ATOMIC_RELAXED, __HIP_MEMORY_SCOPE_AGENT) < tgt)
        __builtin_amdgcn_s_sleep(2);
    }
    __syncthreads();
    __threadfence();
  }
}

// ---------------- final linear: out = hs @ lin_W^T + lin_b ----------------
// M = S*B = 131072, N = O = 128, K = H = 512. One WG per 128 M rows.

__global__ __launch_bounds__(256) void out_gemm(
    const __bf16* __restrict__ hsx, const __bf16* __restrict__ linwb,
    const float* __restrict__ lin_b, float* __restrict__ out)
{
  __bf16* lds = (__bf16*)smem_raw;     // 128 x LP (lin_W, row o = B-column)
  const int tid = threadIdx.x, w = tid >> 5, lane = tid & 31;
  const int n = lane & 15, hi = lane >> 4;
  const size_t row0 = (size_t)blockIdx.x * 128;

  for (int idx = tid; idx < 128 * 64; idx += 256) {
    int row = idx >> 6, ch = idx & 63;
    *(uint4*)(lds + row * LP + ch * 8) =
        *(const uint4*)(linwb + (size_t)row * HD + ch * 8);
  }
  __syncthreads();

  v8f acc[8] = {};
  const __bf16* A = hsx + (size_t)BA * HD;   // skip the h0 slot
  #pragma unroll 4
  for (int kk = 0; kk < HD / 32; ++kk) {
    int kbase = kk * 32 + hi * 16;
    v16bf a = *(const v16bf*)(A + (row0 + (size_t)w * 16 + n) * HD + kbase);
    #pragma unroll
    for (int nt = 0; nt < 8; ++nt) {
      v16bf bf = *(const v16bf*)(lds + (nt * 16 + n) * LP + kbase);
      acc[nt] = __builtin_amdgcn_wmma_f32_16x16x32_bf16(
          false, a, false, bf, (short)0, acc[nt], false, false);
    }
  }
  #pragma unroll
  for (int nt = 0; nt < 8; ++nt) {
    float bb = lin_b[nt * 16 + n];
    #pragma unroll
    for (int r = 0; r < 8; ++r)
      out[(row0 + (size_t)w * 16 + r + 8 * hi) * OD + nt * 16 + n] = acc[nt][r] + bb;
  }
}

// ---------------- host side ----------------

extern "C" void kernel_launch(void* const* d_in, const int* in_sizes, int n_in,
                              void* d_out, int out_size, void* d_ws, size_t ws_size,
                              hipStream_t stream) {
  (void)in_sizes; (void)n_in; (void)out_size; (void)ws_size;

  const float* C      = (const float*)d_in[0];
  const float* t      = (const float*)d_in[1];
  /* d_in[2] = mask, unused by the reference */
  const float* W_ih   = (const float*)d_in[3];
  const float* W_hh   = (const float*)d_in[4];
  const float* b_ih   = (const float*)d_in[5];
  const float* b_hh   = (const float*)d_in[6];
  const float* lin_W  = (const float*)d_in[7];
  const float* lin_b  = (const float*)d_in[8];
  const float* time_W = (const float*)d_in[9];
  const float* time_b = (const float*)d_in[10];

  // workspace layout (all offsets 256B aligned)
  char* ws = (char*)d_ws;
  constexpr size_t OFF_BAR   = 0;
  constexpr size_t OFF_TDEL  = 256;
  constexpr size_t OFF_U     = OFF_TDEL  + (size_t)SL * BA * 4;
  constexpr size_t OFF_V2    = OFF_U     + (size_t)G4H * 4;
  constexpr size_t OFF_BASE2 = OFF_V2    + (size_t)G4H * 4;
  constexpr size_t OFF_WHHB  = OFF_BASE2 + (size_t)BA * G4H * 4;
  constexpr size_t OFF_LINWB = OFF_WHHB  + (size_t)G4H * HD * 2;
  constexpr size_t OFF_HSX   = OFF_LINWB + (size_t)OD * HD * 2;

  unsigned int* bar   = (unsigned int*)(ws + OFF_BAR);
  float*  tdel  = (float*)(ws + OFF_TDEL);
  float*  u     = (float*)(ws + OFF_U);
  float*  v2    = (float*)(ws + OFF_V2);
  float*  base2 = (float*)(ws + OFF_BASE2);
  __bf16* whhb  = (__bf16*)(ws + OFF_WHHB);
  __bf16* linwb = (__bf16*)(ws + OFF_LINWB);
  __bf16* hsx   = (__bf16*)(ws + OFF_HSX);

  constexpr size_t LDS_BYTES = (size_t)128 * LP * 2;   // 135168 B (< 320KB/WGP)

  zero_h0<<<(BA * HD + 255) / 256, 256, 0, stream>>>(hsx, bar);
  compute_tdelta<<<(SL * BA + 255) / 256, 256, 0, stream>>>(t, tdel);
  compute_uv<<<G4H / 256, 256, 0, stream>>>(W_ih, b_ih, b_hh, time_W, time_b, u, v2);
  convert_bf16<<<(G4H * HD + OD * HD + 255) / 256, 256, 0, stream>>>(W_hh, lin_W, whhb, linwb);
  compute_base2<<<dim3(G4H / 256, BA), 256, 0, stream>>>(C, W_ih, v2, base2);

  lstm_persistent<<<NWG, 256, LDS_BYTES, stream>>>(tdel, base2, u, whhb, hsx, bar);
  out_gemm<<<(SL * BA) / 128, 256, LDS_BYTES, stream>>>(hsx, linwb, lin_b, (float*)d_out);
}
